// StridedwindowAttention_33483565040135
// MI455X (gfx1250) — compile-verified, matrix operand order checked by reading, not run on hardware
//
#include <hip/hip_runtime.h>

// ---------------------------------------------------------------------------
// Strided-window attention, fused per (batch, head, s1, s2) group.
// B=4, C=256, H=W=64, heads=8, head_dim=32, stride=4 -> 512 groups, L=256.
// One workgroup (256 threads = 8 wave32) per group. All GEMMs via
// v_wmma_f32_16x16x32_bf16 (bf16 operands, f32 accumulate).
// ---------------------------------------------------------------------------

typedef __attribute__((ext_vector_type(16))) __bf16 v16bf;
typedef __attribute__((ext_vector_type(8)))  __bf16 v8bf;
typedef __attribute__((ext_vector_type(4)))  __bf16 v4bf;
typedef __attribute__((ext_vector_type(8)))  float  v8f;
typedef __attribute__((ext_vector_type(4)))  float  v4f;

#define STRIDE4 4
#define C    256
#define H    64
#define W    64
#define L    256   // pixels per window
#define DH   32    // head dim
#define HALF 128   // score rows processed per pass

// LDS layout (bytes)
#define SBUF_OFF 0                       // 128*256 f32 = 131072 (aliases X-stage: 256*256 bf16)
#define QBUF_OFF 131072                  // 256*32 bf16 = 16384
#define KBUF_OFF (131072 + 16384)
#define VBUF_OFF (131072 + 2 * 16384)    // transposed: 32 x 256
#define WBUF_OFF (131072 + 3 * 16384)    // weight slice 32 x 256
#define MQ_OFF   (131072 + 4 * 16384)    // 256 f32
#define MK_OFF   (MQ_OFF + 1024)
#define SMEM_BYTES (MK_OFF + 1024)       // 198656 bytes (<320KB WGP LDS)

// A-style bf16 16x16x32 fragment load (also used for B fragments, since every
// B operand here is a transpose of a row-major LDS matrix).
// Lane l<16: row=l,  K = 0..7 and 16..23 ; lane>=16: row=l-16, K = 8..15 and 24..31.
__device__ __forceinline__ v16bf load_frag_bf16(const __bf16* base, int row0, int k0, int ld) {
    const int lane = threadIdx.x & 31;
    const __bf16* p = base + (row0 + (lane & 15)) * ld + k0 + ((lane >> 4) * 8);
    union { v16bf f; v8bf h[2]; } u;
    u.h[0] = *(const v8bf*)(p);        // 16B -> ds_load_b128
    u.h[1] = *(const v8bf*)(p + 16);   // 16B -> ds_load_b128
    return u.f;
}

__device__ __forceinline__ v8f wmma_bf16(v16bf a, v16bf b, v8f c) {
    return __builtin_amdgcn_wmma_f32_16x16x32_bf16(false, a, false, b, (short)0, c, false, false);
}

// Project one head's channels for this window: dst = X(LxC) @ Wh^T(Cx32) + bias,
// results stored as bf16 (optionally transposed as 32xL for the V operand).
__device__ void project_head(const float* __restrict__ fea,
                             const float* __restrict__ wmat,
                             const float* __restrict__ bias,
                             char* smem, __bf16* dst,
                             int b, int h, int s1, int s2,
                             float scale, bool transpose_dst)
{
    __bf16* xstage = (__bf16*)(smem + SBUF_OFF);
    __bf16* wbuf   = (__bf16*)(smem + WBUF_OFF);
    const int tid = threadIdx.x, lane = tid & 31, wave = tid >> 5;

    // Stage activations: thread t owns pixel l=t, converts all 256 channels.
    {
        const int l = tid;
        const int y = (l >> 4) * STRIDE4 + s1;
        const int x = (l & 15) * STRIDE4 + s2;
        const float* src = fea + ((size_t)b * C * H + y) * (size_t)W + x;
        __bf16* drow = xstage + l * C;
        #pragma unroll 8
        for (int c = 0; c < C; ++c)
            drow[c] = (__bf16)src[(size_t)c * H * W];
    }
    // Stage this head's 32x256 weight slice (vectorized, coalesced).
    {
        const float* wsrc = wmat + (size_t)h * DH * C;
        for (int i = tid * 4; i < DH * C; i += 256 * 4) {
            v4f wv = *(const v4f*)(wsrc + i);
            v4bf wb;
            wb[0] = (__bf16)wv[0]; wb[1] = (__bf16)wv[1];
            wb[2] = (__bf16)wv[2]; wb[3] = (__bf16)wv[3];
            *(v4bf*)(wbuf + i) = wb;
        }
    }
    __syncthreads();

    // 16 m-tiles x 2 n-tiles = 32 WMMA tiles over 8 waves; K = 256 in 8 steps.
    for (int it = 0; it < 4; ++it) {
        const int t  = wave + 8 * it;
        const int mt = t >> 1, nt = t & 1;
        v8f acc = {};
        #pragma unroll
        for (int ko = 0; ko < C; ko += 32) {
            v16bf a  = load_frag_bf16(xstage, mt * 16, ko, C);
            v16bf bb = load_frag_bf16(wbuf,   nt * 16, ko, C);
            acc = wmma_bf16(a, bb, acc);
        }
        const int n = lane & 15, hs = lane >> 4;
        const float bv = bias[h * DH + nt * 16 + n];
        #pragma unroll
        for (int r = 0; r < 8; ++r) {
            const int row = mt * 16 + r + 8 * hs;
            const int col = nt * 16 + n;
            const float val = (acc[r] + bv) * scale;
            if (transpose_dst) dst[col * L + row] = (__bf16)val;   // 32 x L
            else               dst[row * DH + col] = (__bf16)val;  // L x 32
        }
    }
    __syncthreads();
}

__global__ __launch_bounds__(256, 1)
void swattn_kernel(const float* __restrict__ fea_q, const float* __restrict__ fea_k,
                   const float* __restrict__ fea_v,
                   const float* __restrict__ mask_q, const float* __restrict__ mask_k,
                   const float* __restrict__ q_w, const float* __restrict__ q_b,
                   const float* __restrict__ k_w, const float* __restrict__ k_b,
                   const float* __restrict__ v_w, const float* __restrict__ v_b,
                   float* __restrict__ out)
{
    extern __shared__ char smem[];
    float*  sbuf = (float*)(smem + SBUF_OFF);    // 128 x 256 f32 scores / bf16 P rows
    __bf16* qbuf = (__bf16*)(smem + QBUF_OFF);   // L x 32
    __bf16* kbuf = (__bf16*)(smem + KBUF_OFF);   // L x 32
    __bf16* vbuf = (__bf16*)(smem + VBUF_OFF);   // 32 x L (transposed)
    float*  mqv  = (float*)(smem + MQ_OFF);
    float*  mkv  = (float*)(smem + MK_OFF);

    const int g  = blockIdx.x;                   // 512 groups
    const int s2 = g & 3, s1 = (g >> 2) & 3, h = (g >> 4) & 7, b = g >> 7;
    const int tid = threadIdx.x, lane = tid & 31, wave = tid >> 5;

    // Stage window masks (one pixel per thread).
    {
        const int l = tid;
        const int y = (l >> 4) * STRIDE4 + s1;
        const int x = (l & 15) * STRIDE4 + s2;
        mqv[l] = mask_q[((size_t)b * H + y) * W + x];
        mkv[l] = mask_k[((size_t)b * H + y) * W + x];
    }

    const float scaling = 0.17677669529663687f;  // head_dim^-0.5, folded into Q
    project_head(fea_q, q_w, q_b, smem, qbuf, b, h, s1, s2, scaling, false);
    project_head(fea_k, k_w, k_b, smem, kbuf, b, h, s1, s2, 1.0f,    false);
    project_head(fea_v, v_w, v_b, smem, vbuf, b, h, s1, s2, 1.0f,    true);

    for (int hb = 0; hb < 2; ++hb) {
        const int r0 = hb * HALF;
        __syncthreads();   // prior pass's P / V-projection reads of sbuf complete

        // ---- S = Q[r0:r0+128] @ K^T : wave owns m-tile 'wave'; K=32 -> 1 WMMA/tile
        {
            v16bf a = load_frag_bf16(qbuf, r0 + wave * 16, 0, DH);
            const int n = lane & 15, hs = lane >> 4;
            for (int nt = 0; nt < 16; ++nt) {
                v16bf bb = load_frag_bf16(kbuf, nt * 16, 0, DH);
                v8f acc = {};
                acc = wmma_bf16(a, bb, acc);
                #pragma unroll
                for (int r = 0; r < 8; ++r)
                    sbuf[(wave * 16 + r + 8 * hs) * 256 + nt * 16 + n] = acc[r];
            }
        }
        __syncthreads();

        // ---- Masked softmax, one wave per row; P written in place as bf16.
        for (int i = 0; i < 16; ++i) {
            const int lr  = wave * 16 + i;
            const float mqr = mqv[r0 + lr];
            float* srow = sbuf + lr * 256;
            const int j0 = lane * 8;
            v4f sa = *(const v4f*)(srow + j0);
            v4f sb = *(const v4f*)(srow + j0 + 4);
            v4f ma = *(const v4f*)(mkv + j0);
            v4f mb = *(const v4f*)(mkv + j0 + 4);
            float vals[8], mk8[8];
            #pragma unroll
            for (int e = 0; e < 8; ++e) {
                vals[e] = (e < 4) ? sa[e] : sb[e - 4];
                mk8[e]  = (e < 4) ? ma[e] : mb[e - 4];
            }
            float mx = -__builtin_inff();
            #pragma unroll
            for (int e = 0; e < 8; ++e)
                if (mk8[e] > 0.5f) mx = fmaxf(mx, vals[e]);
            #pragma unroll
            for (int off = 16; off > 0; off >>= 1)
                mx = fmaxf(mx, __shfl_xor(mx, off, 32));
            float p[8], sum = 0.0f;
            #pragma unroll
            for (int e = 0; e < 8; ++e) {
                const float ev = (mk8[e] > 0.5f) ? __expf(vals[e] - mx) : 0.0f;
                p[e] = ev; sum += ev;
            }
            #pragma unroll
            for (int off = 16; off > 0; off >>= 1)
                sum += __shfl_xor(sum, off, 32);
            // Row zeroed if query pixel masked or no valid keys (reference gate).
            const float gate = (mqr > 0.5f && sum > 0.0f) ? (1.0f / sum) : 0.0f;
            v8bf pb;
            #pragma unroll
            for (int e = 0; e < 8; ++e) pb[e] = (__bf16)(p[e] * gate);
            *(v8bf*)((__bf16*)srow + j0) = pb;   // safe: same-wave LDS ops are in-order
        }
        __syncthreads();

        // ---- O = P @ V : m-tile = wave, two 16-col n-tiles, K = 256.
        {
            const __bf16* pbase = (const __bf16*)sbuf;  // bf16 rows at stride 512 elems
            const int n = lane & 15, hs = lane >> 4;
            for (int nt = 0; nt < 2; ++nt) {
                v8f acc = {};
                #pragma unroll
                for (int ko = 0; ko < 256; ko += 32) {
                    v16bf a  = load_frag_bf16(pbase, wave * 16, ko, 512);
                    v16bf bb = load_frag_bf16(vbuf,  nt * 16,  ko, L);
                    acc = wmma_bf16(a, bb, acc);
                }
                const int d = nt * 16 + n;
                #pragma unroll
                for (int r = 0; r < 8; ++r) {
                    const int lrow = wave * 16 + r + 8 * hs;
                    const int gq = r0 + lrow;                 // window pixel index
                    const int y = (gq >> 4) * STRIDE4 + s1;
                    const int x = (gq & 15) * STRIDE4 + s2;
                    out[(((size_t)b * 256 + h * DH + d) * H + y) * W + x] = acc[r];
                }
            }
        }
    }
}

extern "C" void kernel_launch(void* const* d_in, const int* in_sizes, int n_in,
                              void* d_out, int out_size, void* d_ws, size_t ws_size,
                              hipStream_t stream) {
    (void)in_sizes; (void)n_in; (void)out_size; (void)d_ws; (void)ws_size;
    const float* fea_q  = (const float*)d_in[0];
    const float* fea_k  = (const float*)d_in[1];
    const float* fea_v  = (const float*)d_in[2];
    const float* mask_q = (const float*)d_in[3];
    const float* mask_k = (const float*)d_in[4];
    const float* q_w    = (const float*)d_in[5];
    const float* q_b    = (const float*)d_in[6];
    const float* k_w    = (const float*)d_in[7];
    const float* k_b    = (const float*)d_in[8];
    const float* v_w    = (const float*)d_in[9];
    const float* v_b    = (const float*)d_in[10];
    float* out = (float*)d_out;

    // Opt in to >64KB dynamic LDS (no-op if already allowed; not a stream op,
    // safe and deterministic under graph capture).
    (void)hipFuncSetAttribute((const void*)swattn_kernel,
                              hipFuncAttributeMaxDynamicSharedMemorySize,
                              SMEM_BYTES);

    dim3 grid(512);   // B(4) * heads(8) * stride(4) * stride(4)
    dim3 block(256);  // 8 wave32
    swattn_kernel<<<grid, block, SMEM_BYTES, stream>>>(
        fea_q, fea_k, fea_v, mask_q, mask_k,
        q_w, q_b, k_w, k_b, v_w, v_b, out);
}
